// TransformerWithMoLA_26027501813958
// MI455X (gfx1250) — matrix operand order repaired and checked by reading, not compiled, not run on hardware
//
#include <hip/hip_runtime.h>
#include <hip/hip_bf16.h>

// ---------- types ----------
typedef __attribute__((ext_vector_type(16))) __bf16 v16bf;
typedef __attribute__((ext_vector_type(8)))  __bf16 v8bf;
typedef __attribute__((ext_vector_type(8)))  float  v8f;

#define Bc 8
#define Sc 4096
#define Hc 1024
#define Ec 8
#define Rc 16
#define HQ 256          // H/4
#define TOK (Bc*Sc)     // 32768

// per-wave LDS layout (bytes)
#define XL_OFF   0            // 16*1024 bf16 = 32768
#define HL_OFF   32768        // 16*256  bf16 = 8192
#define LL_OFF   40960        // 16*8    f32  = 512
#define GW_OFF   41472        // 16*2    f32  = 128
#define AL_OFF   41600        // 16*32   bf16 = 1024
#define WAVE_LDS 42624
#define WAVES_PER_BLK 4
#define SMEM_BYTES (WAVE_LDS * WAVES_PER_BLK)

// ---------- helpers ----------
static __device__ inline unsigned short f2bfbits(float f) {
    unsigned u = __builtin_bit_cast(unsigned, f);
    unsigned r = u + 0x7FFFu + ((u >> 16) & 1u);   // round-to-nearest-even
    return (unsigned short)(r >> 16);
}
static __device__ inline __bf16 f2bf(float f) {
    unsigned short h = f2bfbits(f);
    return __builtin_bit_cast(__bf16, h);
}
static __device__ inline float bf2f(__bf16 b) {
    unsigned u = ((unsigned)__builtin_bit_cast(unsigned short, b)) << 16;
    return __builtin_bit_cast(float, u);
}
static __device__ inline unsigned pack2(float a, float b) {
    return ((unsigned)f2bfbits(b) << 16) | (unsigned)f2bfbits(a);
}
static __device__ inline v16bf cat16(v8bf lo, v8bf hi) {
    return __builtin_shufflevector(lo, hi, 0,1,2,3,4,5,6,7,8,9,10,11,12,13,14,15);
}
static __device__ inline v8f wmma_bf16(v16bf a, v16bf b, v8f c) {
    // D = A(16x32) * B(32x16) + C, f32 accum
    return __builtin_amdgcn_wmma_f32_16x16x32_bf16(false, a, false, b, (short)0, c, false, false);
}

// ---------- weight pre-conversion: fp32 -> bf16 (runs once per launch) ----------
__global__ void __launch_bounds__(256)
cvt_weights(const float* __restrict__ W1, const float* __restrict__ A,
            const float* __restrict__ Bw,
            __bf16* __restrict__ W1b, __bf16* __restrict__ Ab,
            __bf16* __restrict__ Bwb) {
    const int N1 = HQ * Hc;        // 262144
    const int N2 = Ec * Rc * Hc;   // 131072
    const int N3 = Ec * Hc * Rc;   // 131072
    int gid = blockIdx.x * 256 + threadIdx.x;
    if (gid < N1)                W1b[gid]           = f2bf(W1[gid]);
    else if (gid < N1 + N2)      Ab[gid - N1]       = f2bf(A[gid - N1]);
    else if (gid < N1 + N2 + N3) Bwb[gid - N1 - N2] = f2bf(Bw[gid - N1 - N2]);
}

// ---------- fused MoLA kernel: one wave == one 16-token tile ----------
__global__ void __launch_bounds__(128)
mola_fused(const float* __restrict__ x, const float* __restrict__ b1,
           const float* __restrict__ W2, const float* __restrict__ b2,
           const int* __restrict__ fids,
           const __bf16* __restrict__ W1b, const __bf16* __restrict__ Ab,
           const __bf16* __restrict__ Bwb, float* __restrict__ out) {
    extern __shared__ __align__(16) char smem[];
    const int lane = threadIdx.x & 31;
    const int widx = threadIdx.x >> 5;
    const int tok0 = (blockIdx.x * WAVES_PER_BLK + widx) * 16;
    const int bb   = tok0 / Sc;            // whole tile is inside one batch row

    char*   base = smem + widx * WAVE_LDS;
    __bf16* xl   = (__bf16*)(base + XL_OFF);
    __bf16* hl   = (__bf16*)(base + HL_OFF);
    float*  ll   = (float*) (base + LL_OFF);
    float*  gwl  = (float*) (base + GW_OFF);
    __bf16* al   = (__bf16*)(base + AL_OFF);

    // ---- stage x tile (16 x 1024) into LDS as bf16, read x from HBM exactly once ----
    {
        const float4* xg = (const float4*)(x + (size_t)tok0 * Hc);
        unsigned* xlu = (unsigned*)xl;
        for (int i = lane; i < 16 * Hc / 4; i += 32) {
            float4 v = xg[i];
            xlu[i * 2 + 0] = pack2(v.x, v.y);
            xlu[i * 2 + 1] = pack2(v.z, v.w);
        }
    }
    __syncthreads();

    // lane-role constants; A and B fragments both use contiguous k = 16*half + s
    const int  rlo = lane & 15;
    const bool hiL = lane >= 16;
    const int  c0  = hiL ? 8 : 0;     // unused; A-frag K also uses kb
    const int  kb  = hiL ? 16 : 0;    // B-frag K half select

    // ---- phase 1: gate GEMM  h = relu(x @ W1^T + b1)  (16 N-tiles, K=1024) ----
    for (int nt = 0; nt < 16; ++nt) {
        v8f acc = {};
        const int f = nt * 16 + rlo;
        const __bf16* wrow = W1b + (size_t)f * Hc;     // W1 row f == B column f
        for (int kk = 0; kk < Hc; kk += 32) {
            v8bf a0 = *(const v8bf*)(xl + rlo * Hc + kk + kb);
            v8bf a1 = *(const v8bf*)(xl + rlo * Hc + kk + kb + 8);
            v16bf bf = *(const v16bf*)(wrow + kk + kb);
            acc = wmma_bf16(cat16(a0, a1), bf, acc);
        }
        const float bias = b1[f];
        #pragma unroll
        for (int v = 0; v < 8; ++v) {
            const int tl = v + (hiL ? 8 : 0);
            float hv = acc[v] + bias;
            hl[tl * HQ + f] = f2bf(hv > 0.f ? hv : 0.f);
        }
    }
    __syncthreads();

    // ---- phase 2: logits (16 tokens x 8 experts, K=256) : 4 pairs per lane ----
    #pragma unroll
    for (int i = 0; i < 4; ++i) {
        const int p = lane * 4 + i;
        const int t = p >> 3, e = p & 7;
        float acc = b2[e];
        const float*  w2r = W2 + e * HQ;
        const __bf16* hr  = hl + t * HQ;
        for (int f = 0; f < HQ; ++f) acc += bf2f(hr[f]) * w2r[f];
        ll[t * 8 + e] = acc;
    }
    __syncthreads();

    // ---- phase 3: masked softmax. Only experts {0, fid+1} survive (exp(-1e9)==0) ----
    const int e1 = fids[bb] + 1;
    if (lane < 16) {
        const int t = lane;
        const float l0 = ll[t * 8 + 0];
        const float l1 = ll[t * 8 + e1];
        const float m  = fmaxf(l0, l1);
        const float p0 = __expf(l0 - m), p1 = __expf(l1 - m);
        const float inv = 2.0f / (p0 + p1);     // SCALING = 2.0 folded in
        gwl[t * 2 + 0] = p0 * inv;
        gwl[t * 2 + 1] = p1 * inv;
    }
    __syncthreads();

    // ---- phase 4: a'[t, j] = gw * (x @ A[e]^T), j = slot*16 + r, bf16 in LDS ----
    for (int slot = 0; slot < 2; ++slot) {
        const int e = slot ? e1 : 0;
        v8f acc = {};
        const __bf16* arow = Ab + (size_t)(e * Rc + rlo) * Hc;  // A[e][r][:] == B column r
        for (int kk = 0; kk < Hc; kk += 32) {
            v8bf a0 = *(const v8bf*)(xl + rlo * Hc + kk + kb);
            v8bf a1 = *(const v8bf*)(xl + rlo * Hc + kk + kb + 8);
            v16bf bf = *(const v16bf*)(arow + kk + kb);
            acc = wmma_bf16(cat16(a0, a1), bf, acc);
        }
        #pragma unroll
        for (int v = 0; v < 8; ++v) {
            const int tl = v + (hiL ? 8 : 0);
            al[tl * 32 + slot * 16 + rlo] = f2bf(acc[v] * gwl[tl * 2 + slot]);
        }
    }
    __syncthreads();

    // ---- phase 5: out[t,h] = a'[t,0:32] @ Bcat[0:32,h], ONE wmma per 16x16 tile ----
    // A'-fragment: fixed for all 64 N-tiles (16 halves/lane from LDS)
    v8bf q0 = *(const v8bf*)(al + rlo * 32 + kb);
    v8bf q1 = *(const v8bf*)(al + rlo * 32 + kb + 8);
    const v16bf apf = cat16(q0, q1);
    // B-fragment: lanes 0-15 carry j=0..15 (expert 0), lanes 16-31 j=16..31 (expert e1);
    // Bw is (E,H,R) so each lane's 16 halves are contiguous: Bwb[e][h][0..15]
    const int eB = hiL ? e1 : 0;
    for (int nt = 0; nt < 64; ++nt) {
        const int hcol = nt * 16 + rlo;
        v16bf bf = *(const v16bf*)(Bwb + ((size_t)eB * Hc + hcol) * Rc);
        v8f acc = {};
        acc = wmma_bf16(apf, bf, acc);
        #pragma unroll
        for (int v = 0; v < 8; ++v) {
            const int tok = tok0 + v + (hiL ? 8 : 0);
            out[(size_t)tok * Hc + nt * 16 + rlo] = acc[v];
        }
    }
}

extern "C" void kernel_launch(void* const* d_in, const int* in_sizes, int n_in,
                              void* d_out, int out_size, void* d_ws, size_t ws_size,
                              hipStream_t stream) {
    const float* x    = (const float*)d_in[0];   // [8,4096,1024]
    const float* A    = (const float*)d_in[1];   // [8,16,1024]
    const float* Bw   = (const float*)d_in[2];   // [8,1024,16]
    const float* W1   = (const float*)d_in[3];   // [256,1024]
    const float* b1   = (const float*)d_in[4];   // [256]
    const float* W2   = (const float*)d_in[5];   // [8,256]
    const float* b2   = (const float*)d_in[6];   // [8]
    const int*   fid  = (const int*)d_in[7];     // [8]
    float* out = (float*)d_out;                  // [8,4096,1024] fp32

    // workspace: bf16 copies of W1 (512KB), A (256KB), Bw (256KB)
    char* ws = (char*)d_ws;
    __bf16* W1b = (__bf16*)(ws);
    __bf16* Ab  = (__bf16*)(ws + 524288);
    __bf16* Bwb = (__bf16*)(ws + 786432);

    {
        const int total = HQ * Hc + Ec * Rc * Hc + Ec * Hc * Rc;  // 524288
        cvt_weights<<<(total + 255) / 256, 256, 0, stream>>>(W1, A, Bw, W1b, Ab, Bwb);
    }
    {
        const int nWaves  = TOK / 16;                   // 2048
        const int nBlocks = nWaves / WAVES_PER_BLK;     // 512
        mola_fused<<<nBlocks, 128, SMEM_BYTES, stream>>>(
            x, b1, W2, b2, fid, W1b, Ab, Bwb, out);
    }
}
